// Int4LinearPacked_19662360281097
// MI455X (gfx1250) — compile-verified, matrix operand order checked by reading, not run on hardware
//
#include <hip/hip_runtime.h>

typedef __attribute__((ext_vector_type(16))) _Float16 v16h;
typedef __attribute__((ext_vector_type(8)))  _Float16 v8h;
typedef __attribute__((ext_vector_type(4)))  _Float16 v4h;
typedef __attribute__((ext_vector_type(8)))  float    v8f;
typedef __attribute__((ext_vector_type(4)))  float    v4f;
typedef __attribute__((ext_vector_type(4)))  int      v4i;

#define IN_F   4096
#define OUT_F  16384
#define M_TOT  8192      // 4 * 2048
#define BM     128       // block tile M
#define BN     256       // block tile N
#define BK     32        // K per stage (one WMMA K-step)
#define LDA    (BK + 8)  // halves, padded against LDS bank conflicts
#define LDB    (BK + 8)

__global__ __launch_bounds__(256)
void int4_linear_wmma(const float* __restrict__ X,
                      const int*   __restrict__ Wp,     // one packed byte per int
                      const float* __restrict__ scale,
                      const float* __restrict__ bias,
                      float* __restrict__ Y)
{
    // double-buffered stages: 2*(128*40 + 256*40)*2B = 60 KB of the 320 KB LDS
    __shared__ _Float16 As[2][BM * LDA];
    __shared__ _Float16 Bs[2][BN * LDB];

    const int tid  = threadIdx.x;
    const int lane = tid & 31;
    const int wave = tid >> 5;
    const int wm   = wave & 1;   // 2 wave rows over M  (64 rows each)
    const int wn   = wave >> 1;  // 4 wave cols over N  (64 cols each)

    const int mTile = blockIdx.y * BM;
    const int nTile = blockIdx.x * BN;

    // ---- global->register staging for the software pipeline ----
    v4f aReg[4];   // A tile: 128x32 f32 = 1024 float4 -> 4 per thread
    v4i bReg[4];   // B tile: 256 rows x 16 bytes = 1024 int4-vecs -> 4 per thread

    const int kIters = IN_F / BK;   // 128

    auto loadRegs = [&](int kt) {
        const int k0 = kt * BK;
        #pragma unroll
        for (int i = 0; i < 4; ++i) {
            int v   = tid + i * 256;
            int row = v >> 3;                     // 0..127
            int c4  = v & 7;
            const float* p = X + (size_t)(mTile + row) * IN_F + k0 + c4 * 4;
            aReg[i] = *(const v4f*)p;
        }
        #pragma unroll
        for (int i = 0; i < 4; ++i) {
            int v   = tid + i * 256;
            int row = v >> 2;                     // 0..255
            int kq  = v & 3;                      // 4-byte group within the 16-byte row chunk
            size_t off = (size_t)(nTile + row) * (IN_F / 2) + (size_t)(k0 / 2) + kq * 4;
            bReg[i] = *(const v4i*)(Wp + off);
        }
    };

    auto storeLds = [&](int buf) {
        #pragma unroll
        for (int i = 0; i < 4; ++i) {
            int v   = tid + i * 256;
            int row = v >> 3;
            int c4  = v & 7;
            v4f f = aReg[i];
            v4h h;
            h[0] = (_Float16)f[0]; h[1] = (_Float16)f[1];
            h[2] = (_Float16)f[2]; h[3] = (_Float16)f[3];
            *(v4h*)&As[buf][row * LDA + c4 * 4] = h;     // 8B-aligned ds_store_b64
        }
        #pragma unroll
        for (int i = 0; i < 4; ++i) {
            int v   = tid + i * 256;
            int row = v >> 2;
            int kq  = v & 3;
            v8h h;
            #pragma unroll
            for (int j = 0; j < 4; ++j) {                // each int = one byte = 2 nibbles
                int w  = bReg[i][j];
                int lo = (w << 28) >> 28;                // even k, branchless sign-extend
                int hi = (w << 24) >> 28;                // odd  k
                h[2 * j]     = (_Float16)lo;
                h[2 * j + 1] = (_Float16)hi;
            }
            *(v8h*)&Bs[buf][row * LDB + kq * 8] = h;     // 16B-aligned ds_store_b128
        }
    };

    // ---- accumulators: 4 (M) x 4 (N) tiles of 16x16 f32 = 128 VGPRs ----
    v8f acc[4][4];
    #pragma unroll
    for (int mi = 0; mi < 4; ++mi)
        #pragma unroll
        for (int ni = 0; ni < 4; ++ni)
            acc[mi][ni] = 0.0f;

    loadRegs(0);
    storeLds(0);
    __syncthreads();

    const int lrow = lane & 15;
    const int koff = (lane & 16) ? 8 : 0;   // ISA 16-bit A/B layout: hi half-wave holds K+8 / K+24

    for (int kt = 0; kt < kIters; ++kt) {
        const int cur = kt & 1;
        if (kt + 1 < kIters) loadRegs(kt + 1);   // global loads overlap the WMMAs below

        // B fragments: this wave's four N sub-tiles
        v16h bfrag[4];
        #pragma unroll
        for (int ni = 0; ni < 4; ++ni) {
            const _Float16* pb = &Bs[cur][(wn * 64 + ni * 16 + lrow) * LDB + koff];
            v8h lo = *(const v8h*)pb;            // K = koff .. koff+7
            v8h hi = *(const v8h*)(pb + 16);     // K = koff+16 .. koff+23
            #pragma unroll
            for (int j = 0; j < 8; ++j) { bfrag[ni][j] = lo[j]; bfrag[ni][j + 8] = hi[j]; }
        }

        #pragma unroll
        for (int mi = 0; mi < 4; ++mi) {
            const _Float16* pa = &As[cur][(wm * 64 + mi * 16 + lrow) * LDA + koff];
            v8h lo = *(const v8h*)pa;
            v8h hi = *(const v8h*)(pa + 16);
            v16h afrag;
            #pragma unroll
            for (int j = 0; j < 8; ++j) { afrag[j] = lo[j]; afrag[j + 8] = hi[j]; }

            #pragma unroll
            for (int ni = 0; ni < 4; ++ni) {
                acc[mi][ni] = __builtin_amdgcn_wmma_f32_16x16x32_f16(
                    false, afrag, false, bfrag[ni],
                    (short)0, acc[mi][ni], false, false);
            }
        }

        if (kt + 1 < kIters) storeLds(cur ^ 1);  // fill the other buffer
        __syncthreads();                          // single barrier per K-step
    }

    // ---- epilogue: y = acc * scale[n] + bias[n], streaming (non-temporal) stores ----
    const int lhi = (lane >> 4) & 1;  // C/D layout: lanes 16-31 hold M rows 8..15
    #pragma unroll
    for (int ni = 0; ni < 4; ++ni) {
        int n = nTile + wn * 64 + ni * 16 + lrow;
        float s = scale[n];
        float b = bias[n];
        #pragma unroll
        for (int mi = 0; mi < 4; ++mi) {
            int mbase = mTile + wm * 64 + mi * 16 + lhi * 8;
            #pragma unroll
            for (int r = 0; r < 8; ++r) {
                float val = acc[mi][ni][r] * s + b;
                __builtin_nontemporal_store(val, &Y[(size_t)(mbase + r) * OUT_F + n]);
            }
        }
    }
}

extern "C" void kernel_launch(void* const* d_in, const int* in_sizes, int n_in,
                              void* d_out, int out_size, void* d_ws, size_t ws_size,
                              hipStream_t stream) {
    (void)in_sizes; (void)n_in; (void)out_size; (void)d_ws; (void)ws_size;
    const float* x     = (const float*)d_in[0];
    const int*   wp    = (const int*)d_in[1];
    const float* scale = (const float*)d_in[2];
    const float* bias  = (const float*)d_in[3];
    float*       y     = (float*)d_out;

    dim3 grid(OUT_F / BN, M_TOT / BM);   // 64 x 64 = 4096 workgroups
    int4_linear_wmma<<<grid, dim3(256), 0, stream>>>(x, wp, scale, bias, y);
}
